// Mamba3Dcross_83442624626914
// MI455X (gfx1250) — matrix operand order repaired and checked by reading, not compiled, not run on hardware
//
#include <hip/hip_runtime.h>
#include <hip/hip_bf16.h>

// ---------------- model constants ----------------
#define T_TOK   32768      // B*H*W*D tokens
#define C_MODEL 128
#define D_INNER 256
#define N_STATE 16
#define DT_RANK 8
#define DBC_PAD 48         // dt_rank + 2*n = 40, padded to 3 WMMA tiles
#define SEQ_L   32
#define N_SEQ   1024

typedef __attribute__((ext_vector_type(16))) _Float16 v16h;
typedef __attribute__((ext_vector_type(8)))  _Float16 v8h;
typedef __attribute__((ext_vector_type(8)))  float    v8f;

// ---------------- small conversion kernels ----------------
__global__ void cvt_f32_to_f16(const float* __restrict__ src,
                               _Float16* __restrict__ dst, int n) {
  int i = blockIdx.x * 256 + threadIdx.x;
  if (i < n) dst[i] = (_Float16)src[i];
}

// convert [srcRows, cols] f32 -> [dstRows, cols] f16, zero-padding extra rows
__global__ void cvt_pad_rows(const float* __restrict__ src,
                             _Float16* __restrict__ dst,
                             int srcRows, int cols, int dstRows) {
  int i = blockIdx.x * 256 + threadIdx.x;
  if (i < dstRows * cols) {
    int r = i / cols;
    dst[i] = (r < srcRows) ? (_Float16)src[i] : (_Float16)0.0f;
  }
}

__global__ void neg_exp_k(const float* __restrict__ src,
                          float* __restrict__ dst, int n) {
  int i = blockIdx.x * 256 + threadIdx.x;
  if (i < n) dst[i] = -__expf(src[i]);
}

// ---------------- WMMA GEMM:  C[M,N] = A[M,K] * W[N,K]^T ----------------
// A, W row-major f16 (K contiguous). One wave -> one 16x16 tile.
// Block = 128 threads (4 waves along M); grid = (M/64, N/16).
__device__ __forceinline__ v8f wmma_tile_f16(const _Float16* __restrict__ A, int lda,
                                             const _Float16* __restrict__ W, int ldw,
                                             int mBase, int nBase, int K, int lane) {
  const int mr = lane & 15;     // row of A / col of B held by this lane
  const int hi = lane >> 4;     // which K-half of the fragment
  const _Float16* arow = A + (size_t)(mBase + mr) * lda;
  const _Float16* brow = W + (size_t)(nBase + mr) * ldw;
  v8f acc = {};
  for (int k0 = 0; k0 < K; k0 += 32) {
    __builtin_prefetch(arow + k0 + 64, 0, 1);
    __builtin_prefetch(brow + k0 + 64, 0, 1);
    // A fragment: lane holds row m, K = k0 + hi*16 .. +15 (contiguous)
    v16h a = *(const v16h*)(arow + k0 + hi * 16);
    // B fragment: lane holds column n, K = k0 + hi*16 .. +15 (contiguous)
    v16h b = *(const v16h*)(brow + k0 + hi * 16);
    acc = __builtin_amdgcn_wmma_f32_16x16x32_f16(
        false, a, false, b, (short)0, acc, false, false);
  }
  return acc;
}

__global__ void gemm_wmma_f16out(const _Float16* __restrict__ A, int lda,
                                 const _Float16* __restrict__ W, int ldw,
                                 _Float16* __restrict__ C, int ldc, int colOff,
                                 int K) {
  const int lane  = threadIdx.x & 31;
  const int wave  = threadIdx.x >> 5;
  const int mBase = (blockIdx.x * 4 + wave) * 16;
  const int nBase = blockIdx.y * 16;
  v8f acc = wmma_tile_f16(A, lda, W, ldw, mBase, nBase, K, lane);
  const int hi  = lane >> 4;
  const int col = colOff + nBase + (lane & 15);
#pragma unroll
  for (int r = 0; r < 8; ++r) {
    int row = mBase + r + hi * 8;
    C[(size_t)row * ldc + col] = (_Float16)acc[r];
  }
}

__global__ void gemm_wmma_f32out(const _Float16* __restrict__ A, int lda,
                                 const _Float16* __restrict__ W, int ldw,
                                 float* __restrict__ C, int ldc, int colOff,
                                 int K, const float* __restrict__ bias) {
  const int lane  = threadIdx.x & 31;
  const int wave  = threadIdx.x >> 5;
  const int mBase = (blockIdx.x * 4 + wave) * 16;
  const int nBase = blockIdx.y * 16;
  v8f acc = wmma_tile_f16(A, lda, W, ldw, mBase, nBase, K, lane);
  const int hi  = lane >> 4;
  const int col = colOff + nBase + (lane & 15);
  const float bv = bias ? bias[col] : 0.0f;
#pragma unroll
  for (int r = 0; r < 8; ++r) {
    int row = mBase + r + hi * 8;
    C[(size_t)row * ldc + col] = acc[r] + bv;
  }
}

// ---------------- depthwise causal conv (d_conv=4) + SiLU ----------------
// one block = one sequence (256 threads = channels); xi lives in xz[:, :256]
__global__ void conv_silu(const _Float16* __restrict__ xz,
                          _Float16* __restrict__ xc,
                          const float* __restrict__ convw,
                          const float* __restrict__ convb,
                          int p0, int p1, int strideTok) {
  const int s = blockIdx.x;
  const int c = threadIdx.x;
  const long base = (long)(s >> 5) * p0 + (long)(s & 31) * p1;
  const float w0 = convw[c * 4 + 0], w1 = convw[c * 4 + 1];
  const float w2 = convw[c * 4 + 2], w3 = convw[c * 4 + 3];
  const float b = convb[c];
  float x0 = 0.f, x1 = 0.f, x2 = 0.f;
  for (int l = 0; l < SEQ_L; ++l) {
    long tok = base + (long)l * strideTok;
    float x3 = (float)xz[tok * 512 + c];
    float v = b + w0 * x0 + w1 * x1 + w2 * x2 + w3 * x3;
    v = v / (1.0f + __expf(-v));           // SiLU
    xc[tok * 256 + c] = (_Float16)v;
    x0 = x1; x1 = x2; x2 = x3;
  }
}

// ---------------- dt = softplus(dbc[:, :8] @ dtW^T + dtb) ----------------
__global__ void dt_softplus(const float* __restrict__ dbc,
                            const float* __restrict__ dtW,
                            const float* __restrict__ dtb,
                            float* __restrict__ dt) {
  const long tok = blockIdx.x;
  const int d = threadIdx.x;
  const float* r = dbc + tok * DBC_PAD;
  float acc = dtb[d];
#pragma unroll
  for (int k = 0; k < DT_RANK; ++k) acc += r[k] * dtW[d * DT_RANK + k];
  dt[tok * 256 + d] = (acc > 20.0f) ? acc : log1pf(__expf(acc));
}

// ---------------- selective scan + skip + SiLU gate ----------------
// one block = one sequence; thread = channel; B/C staged in LDS.
__global__ void scan_gate(const float* __restrict__ dbc,
                          const float* __restrict__ dt,
                          const _Float16* __restrict__ xc,
                          const _Float16* __restrict__ xz,
                          const float* __restrict__ Aneg,
                          const float* __restrict__ Dp,
                          _Float16* __restrict__ y,
                          int p0, int p1, int strideTok) {
  __shared__ float sB[SEQ_L][N_STATE];
  __shared__ float sC[SEQ_L][N_STATE];
  const int s = blockIdx.x;
  const int tid = threadIdx.x;
  const long base = (long)(s >> 5) * p0 + (long)(s & 31) * p1;
  for (int i = tid; i < SEQ_L * 2 * N_STATE; i += 256) {
    int l = i >> 5, j = i & 31;
    long tok = base + (long)l * strideTok;
    float v = dbc[tok * DBC_PAD + DT_RANK + j];
    if (j < N_STATE) sB[l][j] = v; else sC[l][j - N_STATE] = v;
  }
  __syncthreads();
  const int d = tid;
  float Ad[N_STATE], h[N_STATE];
#pragma unroll
  for (int n = 0; n < N_STATE; ++n) { Ad[n] = Aneg[d * N_STATE + n]; h[n] = 0.f; }
  const float dp = Dp[d];
  for (int l = 0; l < SEQ_L; ++l) {
    long tok = base + (long)l * strideTok;
    float dtv = dt[tok * 256 + d];
    float u   = (float)xc[tok * 256 + d];
    float du  = dtv * u;
    float yv  = 0.f;
#pragma unroll
    for (int n = 0; n < N_STATE; ++n) {
      float dA = __expf(dtv * Ad[n]);
      h[n] = dA * h[n] + du * sB[l][n];
      yv += h[n] * sC[l][n];
    }
    yv += u * dp;
    float zv = (float)xz[tok * 512 + 256 + d];
    yv *= zv / (1.0f + __expf(-zv));        // y * silu(z)
    y[tok * 256 + d] = (_Float16)yv;
  }
}

// ---------------- host launch ----------------
extern "C" void kernel_launch(void* const* d_in, const int* in_sizes, int n_in,
                              void* d_out, int out_size, void* d_ws, size_t ws_size,
                              hipStream_t stream) {
  const float* x   = (const float*)d_in[0];
  const float* fcW = (const float*)d_in[28];
  const float* fcb = (const float*)d_in[29];
  // per-direction param base index: 1 + 9*t ; order: inW convw convb xpW dtW dtb Alog Dp outW

  // -------- workspace carve (256B aligned) --------
  char* p = (char*)d_ws;
  auto carve = [&](size_t bytes) -> void* {
    void* r = (void*)p; p += (bytes + 255) & ~(size_t)255; return r;
  };
  _Float16* xh     = (_Float16*)carve((size_t)T_TOK * 128 * 2);
  _Float16* inW16  = (_Float16*)carve((size_t)3 * 512 * 128 * 2);
  _Float16* xpW16  = (_Float16*)carve((size_t)3 * DBC_PAD * 256 * 2);
  _Float16* outW16 = (_Float16*)carve((size_t)3 * 128 * 256 * 2);
  _Float16* fcW16  = (_Float16*)carve((size_t)128 * 384 * 2);
  float*    Aneg   = (float*)   carve((size_t)3 * 256 * 16 * 4);
  _Float16* xz     = (_Float16*)carve((size_t)T_TOK * 512 * 2);
  _Float16* xc     = (_Float16*)carve((size_t)T_TOK * 256 * 2);
  float*    dbc    = (float*)   carve((size_t)T_TOK * DBC_PAD * 4);
  float*    dtbuf  = (float*)   carve((size_t)T_TOK * 256 * 4);
  _Float16* yb     = (_Float16*)carve((size_t)T_TOK * 256 * 2);
  _Float16* out3   = (_Float16*)carve((size_t)T_TOK * 384 * 2);

  // -------- one-time conversions (re-done every call: deterministic) --------
  {
    int n = T_TOK * 128;
    cvt_f32_to_f16<<<(n + 255) / 256, 256, 0, stream>>>(x, xh, n);
  }
  for (int t = 0; t < 3; ++t) {
    const float* inW  = (const float*)d_in[1 + 9 * t + 0];
    const float* xpW  = (const float*)d_in[1 + 9 * t + 3];
    const float* Alog = (const float*)d_in[1 + 9 * t + 6];
    const float* outW = (const float*)d_in[1 + 9 * t + 8];
    int n1 = 512 * 128;
    cvt_f32_to_f16<<<(n1 + 255) / 256, 256, 0, stream>>>(inW, inW16 + (size_t)t * n1, n1);
    int n2 = DBC_PAD * 256;
    cvt_pad_rows<<<(n2 + 255) / 256, 256, 0, stream>>>(xpW, xpW16 + (size_t)t * n2,
                                                       40, 256, DBC_PAD);
    int n3 = 128 * 256;
    cvt_f32_to_f16<<<(n3 + 255) / 256, 256, 0, stream>>>(outW, outW16 + (size_t)t * n3, n3);
    int n4 = 256 * 16;
    neg_exp_k<<<(n4 + 255) / 256, 256, 0, stream>>>(Alog, Aneg + (size_t)t * n4, n4);
  }
  {
    int n = 128 * 384;
    cvt_f32_to_f16<<<(n + 255) / 256, 256, 0, stream>>>(fcW, fcW16, n);
  }

  // direction (p0, p1, strideTok): base = (s>>5)*p0 + (s&31)*p1
  const int P0[3] = {32, 1024, 1024};
  const int P1[3] = {1, 1, 32};
  const int ST[3] = {1024, 32, 1};

  for (int t = 0; t < 3; ++t) {
    const float* convw = (const float*)d_in[1 + 9 * t + 1];
    const float* convb = (const float*)d_in[1 + 9 * t + 2];
    const float* dtW   = (const float*)d_in[1 + 9 * t + 4];
    const float* dtb   = (const float*)d_in[1 + 9 * t + 5];
    const float* Dp    = (const float*)d_in[1 + 9 * t + 7];

    // GEMM1: xz[T,512] = xh[T,128] @ inW^T
    gemm_wmma_f16out<<<dim3(T_TOK / 64, 512 / 16), 128, 0, stream>>>(
        xh, 128, inW16 + (size_t)t * 512 * 128, 128, xz, 512, 0, 128);

    // depthwise causal conv + silu
    conv_silu<<<N_SEQ, 256, 0, stream>>>(xz, xc, convw, convb, P0[t], P1[t], ST[t]);

    // dbc[T,48] = xc[T,256] @ xpW^T (padded)
    gemm_wmma_f32out<<<dim3(T_TOK / 64, DBC_PAD / 16), 128, 0, stream>>>(
        xc, 256, xpW16 + (size_t)t * DBC_PAD * 256, 256, dbc, DBC_PAD, 0, 256,
        (const float*)nullptr);

    // dt = softplus(dbc[:, :8] @ dtW^T + dtb)
    dt_softplus<<<T_TOK, 256, 0, stream>>>(dbc, dtW, dtb, dtbuf);

    // selective scan + skip + gate  -> yb (f16)
    scan_gate<<<N_SEQ, 256, 0, stream>>>(dbc, dtbuf, xc, xz,
                                         Aneg + (size_t)t * 256 * 16, Dp, yb,
                                         P0[t], P1[t], ST[t]);

    // GEMM2: out3[:, t*128:(t+1)*128] = yb[T,256] @ outW^T  (f16 out)
    gemm_wmma_f16out<<<dim3(T_TOK / 64, 128 / 16), 128, 0, stream>>>(
        yb, 256, outW16 + (size_t)t * 128 * 256, 256, out3, 384, t * 128, 256);
  }

  // final: out[T,128] = out3[T,384] @ fcW^T + fcb  (f32 out)
  gemm_wmma_f32out<<<dim3(T_TOK / 64, 128 / 16), 128, 0, stream>>>(
      out3, 384, fcW16, 384, (float*)d_out, 128, 0, 384, fcb);
}
